// SelfAttention_32890859552900
// MI455X (gfx1250) — matrix operand order repaired and checked by reading, not compiled, not run on hardware
//
#include <hip/hip_runtime.h>
#include <math.h>

#define Nn    50000
#define Ee    800000
#define HIDc  128
#define Hh    8
#define Dd    16
#define HDc   128
#define FFc   512
#define EPSc  1e-5f
#define SCALEc 0.25f   // 16^-0.5

typedef __attribute__((ext_vector_type(16))) __bf16 v16bf;
typedef __attribute__((ext_vector_type(8)))  __bf16 v8bf;
typedef __attribute__((ext_vector_type(8)))  float  v8f;
typedef unsigned short u16b;
typedef unsigned int   u32b;

// ---------- bf16 helpers ----------
__device__ __forceinline__ __bf16 us2bf(u16b u) {
    return __builtin_bit_cast(__bf16, u);
}
__device__ __forceinline__ u16b f2bfu(float f) {
    u32b x = __builtin_bit_cast(u32b, f);
    u32b r = (x + 0x7FFFu + ((x >> 16) & 1u)) >> 16;   // round-to-nearest-even
    return (u16b)r;
}

// ---------- WMMA fragment loaders (ISA 7.12.2 layouts, wave32) ----------
// A: 16x32 bf16 tile, row-major at (row0,k0), ld elems/row. Each lane needs
// two contiguous 16B groups -> 2x b128 loads, zero per-element shuffling.
// lane<16: M=lane, K {0..7,16..23}; lane>=16: M=lane-16, K {8..15,24..31}
__device__ __forceinline__ v16bf load_a_bf16(const u16b* A, int ld) {
    int lane = threadIdx.x & 31;
    int m    = lane & 15;
    int kb   = (lane >> 4) * 8;
    const u16b* r = A + (size_t)m * ld + kb;
    uint4 lo = *(const uint4*)r;          // K kb..kb+7
    uint4 hi = *(const uint4*)(r + 16);   // K kb+16..kb+23
    v8bf a0 = __builtin_bit_cast(v8bf, lo);
    v8bf a1 = __builtin_bit_cast(v8bf, hi);
    return __builtin_shufflevector(a0, a1, 0, 1, 2, 3, 4, 5, 6, 7,
                                   8, 9, 10, 11, 12, 13, 14, 15);
}
// A tile from f32 source (vector loads, on-the-fly bf16 convert)
__device__ __forceinline__ v16bf load_a_f32(const float* A, int ld) {
    int lane = threadIdx.x & 31;
    int m    = lane & 15;
    int kb   = (lane >> 4) * 8;
    const float* r = A + (size_t)m * ld + kb;
    float4 f0 = *(const float4*)r;
    float4 f1 = *(const float4*)(r + 4);
    float4 f2 = *(const float4*)(r + 16);
    float4 f3 = *(const float4*)(r + 20);
    v16bf f;
    f[0]  = us2bf(f2bfu(f0.x)); f[1]  = us2bf(f2bfu(f0.y));
    f[2]  = us2bf(f2bfu(f0.z)); f[3]  = us2bf(f2bfu(f0.w));
    f[4]  = us2bf(f2bfu(f1.x)); f[5]  = us2bf(f2bfu(f1.y));
    f[6]  = us2bf(f2bfu(f1.z)); f[7]  = us2bf(f2bfu(f1.w));
    f[8]  = us2bf(f2bfu(f2.x)); f[9]  = us2bf(f2bfu(f2.y));
    f[10] = us2bf(f2bfu(f2.z)); f[11] = us2bf(f2bfu(f2.w));
    f[12] = us2bf(f2bfu(f3.x)); f[13] = us2bf(f2bfu(f3.y));
    f[14] = us2bf(f2bfu(f3.z)); f[15] = us2bf(f2bfu(f3.w));
    return f;
}
// B: pre-packed fragment layout. P points at a packed weight
// [(kt*ntn + nt)*32 + lane]*16 contiguous bf16 per lane -> 2x b128 loads.
__device__ __forceinline__ v16bf load_b_pk(const u16b* P, int kt, int nt, int ntn) {
    int lane = threadIdx.x & 31;
    const u16b* src = P + (((size_t)kt * ntn + nt) * 32 + lane) * 16;
    uint4 lo = *(const uint4*)src;
    uint4 hi = *(const uint4*)(src + 8);
    v8bf b0 = __builtin_bit_cast(v8bf, lo);
    v8bf b1 = __builtin_bit_cast(v8bf, hi);
    return __builtin_shufflevector(b0, b1, 0, 1, 2, 3, 4, 5, 6, 7,
                                   8, 9, 10, 11, 12, 13, 14, 15);
}
__device__ __forceinline__ v8f wmma_bf16(v16bf a, v16bf b, v8f c) {
    return __builtin_amdgcn_wmma_f32_16x16x32_bf16(false, a, false, b, (short)0, c,
                                                   false, false);
}

// ---------- float atomic max via CAS ----------
__device__ __forceinline__ void atomicMaxF(float* addr, float val) {
    u32b* a   = (u32b*)addr;
    u32b  cur = *a;
    while (__builtin_bit_cast(float, cur) < val) {
        u32b prev = atomicCAS(a, cur, __builtin_bit_cast(u32b, val));
        if (prev == cur) break;
        cur = prev;
    }
}

// ========================================================================
// K0: pack f32 weight [K x Ncols] into WMMA-B fragment order (bf16).
// One thread per (tile, lane); per-lane 16 contiguous bf16 written.
// B-frag layout: lane<16 -> N=nt*16+lane, K kt*32+{0..7,16..23} ;
//                lane>=16 -> N=nt*16+lane-16, K kt*32+{8..15,24..31}
// ========================================================================
__global__ void packB_kernel(const float* __restrict__ W, u16b* __restrict__ P,
                             int K, int Ncols) {
    int idx  = blockIdx.x * blockDim.x + threadIdx.x;
    int lane = idx & 31;
    int tile = idx >> 5;
    int ntn  = Ncols >> 4;
    int ntk  = K >> 5;
    if (tile >= ntn * ntk) return;
    int kt = tile / ntn, nt = tile % ntn;
    int n  = nt * 16 + (lane & 15);
    int k0 = kt * 32 + (lane >> 4) * 8;
    u16b* dst = P + ((size_t)tile * 32 + lane) * 16;
#pragma unroll
    for (int i = 0; i < 16; ++i)
        dst[i] = f2bfu(W[(size_t)(k0 + (i & 7) + (i >> 3) * 16) * Ncols + n]);
}

// concat bias for fused q|k|v|s projection: [bq, 0, bv, bs]
__global__ void bias512_kernel(const float* __restrict__ bq, const float* __restrict__ bv,
                               const float* __restrict__ bs, float* __restrict__ b512) {
    int i = blockIdx.x * blockDim.x + threadIdx.x;
    if (i >= 4 * HDc) return;
    float v = 0.f;
    if (i < 128)      v = bq[i];
    else if (i < 256) v = 0.f;
    else if (i < 384) v = bv[i - 256];
    else              v = bs[i - 384];
    b512[i] = v;
}

// ========================================================================
// K0b: init scratch (agg=0, denom=0, m=-inf) — required every replay
// ========================================================================
__global__ void init_kernel(float* __restrict__ agg, float* __restrict__ mmax,
                            float* __restrict__ denom) {
    int i = blockIdx.x * blockDim.x + threadIdx.x;
    if (i < Nn * HDc) agg[i] = 0.f;
    if (i < Nn * Hh) { mmax[i] = -INFINITY; denom[i] = 0.f; }
}

// ========================================================================
// K1: shared pre-norm: h = LN(x, pre) [bf16], hff = LN(x, ffpre) [bf16]
// one wave per row, 8 rows per 256-thread block
// ========================================================================
__global__ void prenorm_kernel(const float* __restrict__ x,
                               const float* __restrict__ w0, const float* __restrict__ b0,
                               const float* __restrict__ w1, const float* __restrict__ b1v,
                               u16b* __restrict__ h16, u16b* __restrict__ hff16) {
    int row  = blockIdx.x * 8 + (threadIdx.x >> 5);
    int lane = threadIdx.x & 31;
    if (row >= Nn) return;
    const float* xr = x + (size_t)row * HIDc;
    float v[4];
    float s = 0.f;
#pragma unroll
    for (int i = 0; i < 4; ++i) { v[i] = xr[lane + 32 * i]; s += v[i]; }
#pragma unroll
    for (int off = 16; off >= 1; off >>= 1) s += __shfl_xor(s, off, 32);
    float mean = s * (1.f / HIDc);
    float vs = 0.f;
#pragma unroll
    for (int i = 0; i < 4; ++i) { float d = v[i] - mean; vs += d * d; }
#pragma unroll
    for (int off = 16; off >= 1; off >>= 1) vs += __shfl_xor(vs, off, 32);
    float rstd = rsqrtf(vs * (1.f / HIDc) + EPSc);
#pragma unroll
    for (int i = 0; i < 4; ++i) {
        int   c   = lane + 32 * i;
        float nrm = (v[i] - mean) * rstd;
        h16[(size_t)row * HIDc + c]   = f2bfu(nrm * w0[c] + b0[c]);
        hff16[(size_t)row * HIDc + c] = f2bfu(nrm * w1[c] + b1v[c]);
    }
}

// ========================================================================
// K2: qkvs[N,512] = h @ [Wq|Wk|Wv|Ws] + [bq|0|bv|bs] via WMMA.
// 1 block = 16 rows, 8 waves, each wave 4 of 32 column-tiles.
// Packed weight buffer: 4 matrices, each (4 kt x 8 nt) tiles of 512 elems.
// ========================================================================
__global__ void qkvs_kernel(const u16b* __restrict__ h16, const u16b* __restrict__ wpk,
                            const float* __restrict__ b512, float* __restrict__ qkvs) {
    int tile_m0 = blockIdx.x * 16;
    int wave    = threadIdx.x >> 5;
    int lane    = threadIdx.x & 31;
#pragma unroll
    for (int t = 0; t < 4; ++t) {
        int tile = wave * 4 + t;          // 0..31
        int mat  = tile >> 3;             // which weight matrix
        int nt   = tile & 7;
        const u16b* wbase = wpk + (size_t)mat * (4 * 8 * 512);
        v8f acc = {};
#pragma unroll
        for (int ks = 0; ks < 4; ++ks) {
            v16bf a = load_a_bf16(h16 + (size_t)tile_m0 * HIDc + ks * 32, HIDc);
            v16bf b = load_b_pk(wbase, ks, nt, 8);
            acc = wmma_bf16(a, b, acc);
        }
        int   n    = tile * 16 + (lane & 15);  // 0..511
        int   mb   = (lane >> 4) * 8;
        float bias = b512[n];
#pragma unroll
        for (int r = 0; r < 8; ++r)
            qkvs[(size_t)(tile_m0 + mb + r) * 512 + n] = acc[r] + bias;
    }
}

// ========================================================================
// K3/K4/K5: edge passes, one thread per (edge, head)
// qkvs layout per row: [q(0..127) | k(128..255) | v(256..383) | s(384..511)]
// ========================================================================
__global__ void edge_sim_kernel(const long long* __restrict__ ei,
                                const float* __restrict__ qkvs,
                                float* __restrict__ sim, float* __restrict__ mmax) {
    long long idx = (long long)blockIdx.x * blockDim.x + threadIdx.x;
    if (idx >= (long long)Ee * Hh) return;
    int e   = (int)(idx >> 3);
    int hh  = (int)(idx & 7);
    int src = (int)ei[e];
    int dst = (int)ei[Ee + e];
    const float4* qv = (const float4*)(qkvs + (size_t)dst * 512 + hh * Dd);
    const float4* kv = (const float4*)(qkvs + (size_t)src * 512 + 128 + hh * Dd);
    float s = 0.f;
#pragma unroll
    for (int i = 0; i < 4; ++i) {
        float4 a = qv[i], b = kv[i];
        s += a.x * b.x + a.y * b.y + a.z * b.z + a.w * b.w;
    }
    s *= SCALEc;
    sim[(size_t)e * Hh + hh] = s;
    atomicMaxF(mmax + (size_t)dst * Hh + hh, s);
}

__global__ void edge_exp_kernel(const long long* __restrict__ ei,
                                const float* __restrict__ mmax,
                                float* __restrict__ sim, float* __restrict__ denom) {
    long long idx = (long long)blockIdx.x * blockDim.x + threadIdx.x;
    if (idx >= (long long)Ee * Hh) return;
    int e   = (int)(idx >> 3);
    int hh  = (int)(idx & 7);
    int dst = (int)ei[Ee + e];
    float ev = expf(sim[(size_t)e * Hh + hh] - mmax[(size_t)dst * Hh + hh]);
    sim[(size_t)e * Hh + hh] = ev;
    atomicAdd(denom + (size_t)dst * Hh + hh, ev);
}

__global__ void edge_agg_kernel(const long long* __restrict__ ei,
                                const float* __restrict__ qkvs,
                                const float* __restrict__ sim,
                                const float* __restrict__ denom,
                                float* __restrict__ agg) {
    long long idx = (long long)blockIdx.x * blockDim.x + threadIdx.x;
    if (idx >= (long long)Ee * Hh) return;
    int e   = (int)(idx >> 3);
    int hh  = (int)(idx & 7);
    int src = (int)ei[e];
    int dst = (int)ei[Ee + e];
    float attn = sim[(size_t)e * Hh + hh] / denom[(size_t)dst * Hh + hh];
    const float4* vv = (const float4*)(qkvs + (size_t)src * 512 + 256 + hh * Dd);
    float* ap = agg + (size_t)dst * HDc + hh * Dd;
#pragma unroll
    for (int i = 0; i < 4; ++i) {
        float4 m = vv[i];
        atomicAdd(ap + 4 * i + 0, m.x * attn);
        atomicAdd(ap + 4 * i + 1, m.y * attn);
        atomicAdd(ap + 4 * i + 2, m.z * attn);
        atomicAdd(ap + 4 * i + 3, m.w * attn);
    }
}

// ========================================================================
// K6: gate + output projection + post-LN residual. 1 block = 16 rows,
// 4 waves (128 threads). g = sigmoid([agg|h]@Wg+bg); upd = agg+g*(s-agg);
// ao = upd@Wo+bo; x1 = x + LN(ao, post)
// ========================================================================
__global__ void gated_kernel(const float* __restrict__ x, const float* __restrict__ agg,
                             const float* __restrict__ qkvs, const u16b* __restrict__ h16,
                             const u16b* __restrict__ wgpk, const float* __restrict__ bgp,
                             const u16b* __restrict__ wopk, const float* __restrict__ bop,
                             const float* __restrict__ lnw, const float* __restrict__ lnb,
                             float* __restrict__ x1) {
    __shared__ u16b  updL[16 * HDc];
    __shared__ float aoL[16 * HIDc];
    __shared__ float muL[16], rsL[16];
    int tile_m0 = blockIdx.x * 16;
    int wave = threadIdx.x >> 5, lane = threadIdx.x & 31;
    // stage A: gate GEMM (K=256: agg rows 0..127 of Wg, then h rows 128..255)
#pragma unroll
    for (int t = 0; t < 2; ++t) {
        int ct = wave * 2 + t, col0 = ct * 16;
        v8f acc = {};
#pragma unroll
        for (int ks = 0; ks < 8; ++ks) {
            v16bf a = (ks < 4)
                ? load_a_f32(agg + (size_t)tile_m0 * HDc + ks * 32, HDc)
                : load_a_bf16(h16 + (size_t)tile_m0 * HIDc + (ks - 4) * 32, HIDc);
            v16bf b = load_b_pk(wgpk, ks, ct, 8);
            acc = wmma_bf16(a, b, acc);
        }
        int n = col0 + (lane & 15), mb = (lane >> 4) * 8;
#pragma unroll
        for (int r = 0; r < 8; ++r) {
            int   m  = mb + r, gr = tile_m0 + m;
            float g  = 1.f / (1.f + expf(-(acc[r] + bgp[n])));
            float av = agg[(size_t)gr * HDc + n];
            float sv = qkvs[(size_t)gr * 512 + 384 + n];   // s projection
            updL[m * HDc + n] = f2bfu(av + g * (sv - av));
        }
    }
    __syncthreads();
    // stage B: output projection from LDS upd
#pragma unroll
    for (int t = 0; t < 2; ++t) {
        int ct = wave * 2 + t, col0 = ct * 16;
        v8f acc = {};
#pragma unroll
        for (int ks = 0; ks < 4; ++ks) {
            v16bf a = load_a_bf16(updL + ks * 32, HDc);
            v16bf b = load_b_pk(wopk, ks, ct, 8);
            acc = wmma_bf16(a, b, acc);
        }
        int n = col0 + (lane & 15), mb = (lane >> 4) * 8;
#pragma unroll
        for (int r = 0; r < 8; ++r) aoL[(mb + r) * HIDc + n] = acc[r] + bop[n];
    }
    __syncthreads();
    if (threadIdx.x < 16) {
        int m = threadIdx.x;
        float s = 0.f;
        for (int c = 0; c < HIDc; ++c) s += aoL[m * HIDc + c];
        float mean = s * (1.f / HIDc);
        float vs = 0.f;
        for (int c = 0; c < HIDc; ++c) { float d = aoL[m * HIDc + c] - mean; vs += d * d; }
        muL[m] = mean;
        rsL[m] = rsqrtf(vs * (1.f / HIDc) + EPSc);
    }
    __syncthreads();
    int c = threadIdx.x;  // 0..127
    for (int m = 0; m < 16; ++m) {
        int   gr  = tile_m0 + m;
        float val = (aoL[m * HIDc + c] - muL[m]) * rsL[m] * lnw[c] + lnb[c];
        x1[(size_t)gr * HIDc + c] = x[(size_t)gr * HIDc + c] + val;
    }
}

// ========================================================================
// K7: FFN: out = x1 + LN(relu(hff@W1+b1)@W2+b2, ffpost). 1 block = 16 rows,
// 4 waves, 16KB bf16 activation tile in LDS.
// ========================================================================
__global__ void ffn_kernel(const u16b* __restrict__ hff16, const u16b* __restrict__ w1pk,
                           const float* __restrict__ b1p, const u16b* __restrict__ w2pk,
                           const float* __restrict__ b2p, const float* __restrict__ lnw,
                           const float* __restrict__ lnb, const float* __restrict__ x1,
                           float* __restrict__ out) {
    __shared__ u16b  actL[16 * FFc];
    __shared__ float outL[16 * HIDc];
    __shared__ float muL[16], rsL[16];
    int tile_m0 = blockIdx.x * 16;
    int wave = threadIdx.x >> 5, lane = threadIdx.x & 31;
    // stage 1: act = relu(hff @ W1 + b1), bf16 into LDS (W1: K=128, N=512)
#pragma unroll
    for (int t = 0; t < 8; ++t) {
        int ct = wave * 8 + t, col0 = ct * 16;
        v8f acc = {};
#pragma unroll
        for (int ks = 0; ks < 4; ++ks) {
            v16bf a = load_a_bf16(hff16 + (size_t)tile_m0 * HIDc + ks * 32, HIDc);
            v16bf b = load_b_pk(w1pk, ks, ct, 32);
            acc = wmma_bf16(a, b, acc);
        }
        int n = col0 + (lane & 15), mb = (lane >> 4) * 8;
#pragma unroll
        for (int r = 0; r < 8; ++r)
            actL[(mb + r) * FFc + n] = f2bfu(fmaxf(acc[r] + b1p[n], 0.f));
    }
    __syncthreads();
    // stage 2: ff = act @ W2 + b2 (W2: K=512, N=128)
#pragma unroll
    for (int t = 0; t < 2; ++t) {
        int ct = wave * 2 + t, col0 = ct * 16;
        v8f acc = {};
#pragma unroll
        for (int ks = 0; ks < 16; ++ks) {
            v16bf a = load_a_bf16(actL + ks * 32, FFc);
            v16bf b = load_b_pk(w2pk, ks, ct, 8);
            acc = wmma_bf16(a, b, acc);
        }
        int n = col0 + (lane & 15), mb = (lane >> 4) * 8;
#pragma unroll
        for (int r = 0; r < 8; ++r) outL[(mb + r) * HIDc + n] = acc[r] + b2p[n];
    }
    __syncthreads();
    if (threadIdx.x < 16) {
        int m = threadIdx.x;
        float s = 0.f;
        for (int c = 0; c < HIDc; ++c) s += outL[m * HIDc + c];
        float mean = s * (1.f / HIDc);
        float vs = 0.f;
        for (int c = 0; c < HIDc; ++c) { float d = outL[m * HIDc + c] - mean; vs += d * d; }
        muL[m] = mean;
        rsL[m] = rsqrtf(vs * (1.f / HIDc) + EPSc);
    }
    __syncthreads();
    int c = threadIdx.x;
    for (int m = 0; m < 16; ++m) {
        int   gr  = tile_m0 + m;
        float val = (outL[m * HIDc + c] - muL[m]) * rsL[m] * lnw[c] + lnb[c];
        out[(size_t)gr * HIDc + c] = x1[(size_t)gr * HIDc + c] + val;
    }
}

// ========================================================================
extern "C" void kernel_launch(void* const* d_in, const int* in_sizes, int n_in,
                              void* d_out, int out_size, void* d_ws, size_t ws_size,
                              hipStream_t stream) {
    const float*     x    = (const float*)d_in[0];
    const long long* ei   = (const long long*)d_in[1];
    const float*     Wq   = (const float*)d_in[2];
    const float*     bq   = (const float*)d_in[3];
    const float*     Wk   = (const float*)d_in[4];
    const float*     Wv   = (const float*)d_in[5];
    const float*     bv   = (const float*)d_in[6];
    const float*     Ws   = (const float*)d_in[7];
    const float*     bs   = (const float*)d_in[8];
    const float*     Wg   = (const float*)d_in[9];
    const float*     bg   = (const float*)d_in[10];
    const float*     Wo   = (const float*)d_in[11];
    const float*     bo   = (const float*)d_in[12];
    const float*     W1   = (const float*)d_in[13];
    const float*     b1   = (const float*)d_in[14];
    const float*     W2   = (const float*)d_in[15];
    const float*     b2   = (const float*)d_in[16];
    const float*     lnpw = (const float*)d_in[17];
    const float*     lnpb = (const float*)d_in[18];
    const float*     lnow = (const float*)d_in[19];
    const float*     lnob = (const float*)d_in[20];
    const float*     lnfw = (const float*)d_in[21];
    const float*     lnfb = (const float*)d_in[22];
    const float*     lngw = (const float*)d_in[23];
    const float*     lngb = (const float*)d_in[24];
    float*           out  = (float*)d_out;

    // ---- workspace layout ----
    char* p = (char*)d_ws;
    auto alloc = [&](size_t bytes) -> char* {
        char* r = p;
        p += (bytes + 255) & ~(size_t)255;
        return r;
    };
    u16b*  h16   = (u16b*)alloc((size_t)Nn * HIDc * 2);
    u16b*  hff16 = (u16b*)alloc((size_t)Nn * HIDc * 2);
    float* qkvs  = (float*)alloc((size_t)Nn * 512 * 4);
    float* sim   = (float*)alloc((size_t)Ee * Hh * 4);
    float* mmax  = (float*)alloc((size_t)Nn * Hh * 4);
    float* denom = (float*)alloc((size_t)Nn * Hh * 4);
    float* agg   = (float*)alloc((size_t)Nn * HDc * 4);
    float* x1    = (float*)alloc((size_t)Nn * HIDc * 4);
    // packed weights (fragment order)
    u16b*  wqkvs_pk = (u16b*)alloc((size_t)4 * HIDc * HDc * 2);   // 4 x (4kt x 8nt)
    u16b*  wg_pk    = (u16b*)alloc((size_t)(HDc + HIDc) * HDc * 2);
    u16b*  wo_pk    = (u16b*)alloc((size_t)HDc * HIDc * 2);
    u16b*  w1_pk    = (u16b*)alloc((size_t)HIDc * FFc * 2);
    u16b*  w2_pk    = (u16b*)alloc((size_t)FFc * HIDc * 2);
    float* b512     = (float*)alloc((size_t)4 * HDc * 4);

    const int TB = 256;
    // ---- one-shot weight packing (fragment layout) ----
    // threads per pack = tiles*32 = K*Ncols/16
    packB_kernel<<<(HIDc * HDc / 16 + TB - 1) / TB, TB, 0, stream>>>(
        Wq, wqkvs_pk + 0 * (size_t)HIDc * HDc, HIDc, HDc);
    packB_kernel<<<(HIDc * HDc / 16 + TB - 1) / TB, TB, 0, stream>>>(
        Wk, wqkvs_pk + 1 * (size_t)HIDc * HDc, HIDc, HDc);
    packB_kernel<<<(HIDc * HDc / 16 + TB - 1) / TB, TB, 0, stream>>>(
        Wv, wqkvs_pk + 2 * (size_t)HIDc * HDc, HIDc, HDc);
    packB_kernel<<<(HIDc * HDc / 16 + TB - 1) / TB, TB, 0, stream>>>(
        Ws, wqkvs_pk + 3 * (size_t)HIDc * HDc, HIDc, HDc);
    packB_kernel<<<((HDc + HIDc) * HDc / 16 + TB - 1) / TB, TB, 0, stream>>>(
        Wg, wg_pk, HDc + HIDc, HDc);
    packB_kernel<<<(HDc * HIDc / 16 + TB - 1) / TB, TB, 0, stream>>>(Wo, wo_pk, HDc, HIDc);
    packB_kernel<<<(HIDc * FFc / 16 + TB - 1) / TB, TB, 0, stream>>>(W1, w1_pk, HIDc, FFc);
    packB_kernel<<<(FFc * HIDc / 16 + TB - 1) / TB, TB, 0, stream>>>(W2, w2_pk, FFc, HIDc);
    bias512_kernel<<<2, TB, 0, stream>>>(bq, bv, bs, b512);
    // ---- init scratch (poisoned + graph-replayed, so every call) ----
    init_kernel<<<(Nn * HDc + TB - 1) / TB, TB, 0, stream>>>(agg, mmax, denom);
    // ---- pre-norms ----
    prenorm_kernel<<<Nn / 8, TB, 0, stream>>>(x, lnpw, lnpb, lnfw, lnfb, h16, hff16);
    // ---- fused q/k/v/s projections (WMMA) ----
    qkvs_kernel<<<Nn / 16, TB, 0, stream>>>(h16, wqkvs_pk, b512, qkvs);
    // ---- edge passes ----
    int eblocks = (Ee * Hh) / TB;
    edge_sim_kernel<<<eblocks, TB, 0, stream>>>(ei, qkvs, sim, mmax);
    edge_exp_kernel<<<eblocks, TB, 0, stream>>>(ei, mmax, sim, denom);
    edge_agg_kernel<<<eblocks, TB, 0, stream>>>(ei, qkvs, sim, denom, agg);
    // ---- gated update + out proj + post-LN residual (WMMA) ----
    gated_kernel<<<Nn / 16, 128, 0, stream>>>(x, agg, qkvs, h16, wg_pk, bg, wo_pk, bo,
                                              lnow, lnob, x1);
    // ---- FFN + ffpost-LN residual (WMMA) ----
    ffn_kernel<<<Nn / 16, 128, 0, stream>>>(hff16, w1_pk, b1, w2_pk, b2, lngw, lngb, x1, out);
}
